// EdgeLoss_29326036697507
// MI455X (gfx1250) — matrix operand order repaired and checked by reading, not compiled, not run on hardware
//
#include <hip/hip_runtime.h>

typedef __attribute__((ext_vector_type(16))) _Float16 v16h;
typedef __attribute__((ext_vector_type(8)))  float    v8f;

#define IMG 256

#if __has_builtin(__builtin_amdgcn_global_load_async_to_lds_b32)
#define HAVE_ASYNC_LDS 1
__device__ __forceinline__ void asyncLoadF32(const float* g, float* l) {
  __builtin_amdgcn_global_load_async_to_lds_b32(
      (__attribute__((address_space(1))) int*)(void*)const_cast<float*>(g),
      (__attribute__((address_space(3))) int*)(void*)l, /*offset=*/0, /*cpol=*/0);
}
#else
#define HAVE_ASYNC_LDS 0
#endif

__device__ __forceinline__ void waitAsync() {
#if __has_builtin(__builtin_amdgcn_s_wait_asynccnt)
  __builtin_amdgcn_s_wait_asynccnt(0);
#else
  asm volatile("s_wait_asynccnt 0" ::: "memory");
#endif
}

__device__ __forceinline__ float loadInput(const float* img, int gr, int gc) {
  if (gr < 0 || gr >= IMG || gc < 0 || gc >= IMG) return 0.f;
  return img[gr * IMG + gc];
}

// One block = one 32x32 output tile of one frame, both images.
// Local coord l in [0,64): global row = ty + l - 6, global col = tx + l - 6.
// Gaussian blur on l in [4,51] (3x 16-row WMMA tiles), grad on [5,38],
// output pixels on [6,37].
__global__ __launch_bounds__(256)
void canny_l1_tile_kernel(const float* __restrict__ tgt,
                          const float* __restrict__ mdl,
                          float* __restrict__ partial)
{
  __shared__ __align__(16) float    rawTf[64][67];  // [col][row], f32, transposed
  __shared__ __align__(16) _Float16 outv[48][66];   // vertical-blurred f16, [row-4][col]
  __shared__ float blur[48][49];                    // [row-4][col-4]
  __shared__ float mag [48][49];
  __shared__ float gxs [48][49];
  __shared__ float gys [48][49];
  __shared__ float e0  [32][32];

  float* red = &blur[0][0];                         // reduction aliases dead blur

  const int tid  = threadIdx.x;
  const int lane = tid & 31;
  const int wave = __builtin_amdgcn_readfirstlane(tid >> 5);  // SGPR-uniform
  const int n16  = lane & 15;
  const int half = lane >> 4;

  const int bx    = blockIdx.x;
  const int frame = bx >> 6;
  const int tIdx  = bx & 63;
  const int ty    = (tIdx >> 3) * 32;
  const int tx    = (tIdx & 7) * 32;
  const bool border = (ty == 0) || (ty == IMG - 32) || (tx == 0) || (tx == IMG - 32);

  // 5-tap gaussian, std=1 (matches reference)
  const float g5[5] = {0.054488684549643f, 0.244201342003233f,
                       0.402619946894244f, 0.244201342003233f,
                       0.054488684549643f};

  // Constant WMMA fragments (banded filter matrices), built per-lane.
  // A (16x32 f16): lane m = lane%16; elem (v,j): K = 2*v+j+16*half
  // B (32x16 f16): lane n = lane%16; elem (v,j): K = 2*v+j+16*half
  v16h aG, bG;
  for (int v = 0; v < 8; ++v) {
    for (int j = 0; j < 2; ++j) {
      int ka = 2 * v + j + 16 * half;
      int ta = ka - n16 - 2;
      aG[2 * v + j] = (_Float16)((ta >= 0 && ta < 5) ? g5[ta] : 0.f);
      int kb = 2 * v + j + 16 * half;
      int tb = kb - n16 - 2;
      bG[2 * v + j] = (_Float16)((tb >= 0 && tb < 5) ? g5[tb] : 0.f);
    }
  }

  const int offy[8] = {0, 1, 1, 1, 0, -1, -1, -1};
  const int offx[8] = {1, 1, 0, -1, -1, -1, 0, 1};

  float acc = 0.f;

  for (int imgSel = 0; imgSel < 2; ++imgSel) {
    const float* src = imgSel ? mdl : tgt;

    for (int i = tid; i < 48 * 49; i += 256) {
      int r = i / 49, c = i % 49;
      mag[r][c] = 0.f; gxs[r][c] = 0.f; gys[r][c] = 0.f;
    }
    __syncthreads();

    for (int ch = 0; ch < 3; ++ch) {
      const float* img = src + ((size_t)(frame * 3 + ch)) * (IMG * IMG);

      // ---- stage 0: raw halo tile -> LDS (transposed) ----
#if HAVE_ASYNC_LDS
      if (border) {              // block-uniform: pre-zero so OOB halo stays 0
        for (int i = tid; i < 64 * 64; i += 256)
          rawTf[i >> 6][i & 63] = 0.f;
        __syncthreads();
      }
      for (int i = tid; i < 64 * 64; i += 256) {
        int r = i >> 6, c = i & 63;           // coalesced global, transposed LDS
        int gr = ty + r - 6, gc = tx + c - 6;
        if (gr >= 0 && gr < IMG && gc >= 0 && gc < IMG)
          asyncLoadF32(img + gr * IMG + gc, &rawTf[c][r]);
      }
      waitAsync();
      __syncthreads();
#else
      for (int i = tid; i < 64 * 64; i += 256) {
        int r = i >> 6, c = i & 63;
        rawTf[c][r] = loadInput(img, ty + r - 6, tx + c - 6);
      }
      __syncthreads();
#endif

      // ---- stage 1: vertical gaussian via WMMA (12 tiles, wave-uniform) ----
      for (int t = wave; t < 12; t += 8) {
        int lr0 = 4 + 16 * (t >> 2);   // output row base: 4,20,36
        int lcv = 16 * (t & 3);        // column strip:    0,16,32,48
        const float* colp = &rawTf[lcv + n16][0];
        int base = lr0 - 4 + 16 * half;
        v16h bfrag;
        #pragma unroll
        for (int v = 0; v < 8; ++v) {
          bfrag[2 * v]     = (_Float16)colp[base + 2 * v];
          bfrag[2 * v + 1] = (_Float16)colp[base + 2 * v + 1];
        }
        v8f c = {};
        c = __builtin_amdgcn_wmma_f32_16x16x32_f16(false, aG, false, bfrag,
                                                   (short)0, c, false, false);
        #pragma unroll
        for (int j = 0; j < 8; ++j) {
          int lr = lr0 + j + 8 * half;
          outv[lr - 4][lcv + n16] = (_Float16)c[j];
        }
      }
      __syncthreads();

      // ---- stage 2: horizontal gaussian via WMMA (9 tiles) ----
      for (int p = wave; p < 9; p += 8) {
        int lr0 = 4 + 16 * (p / 3);
        int lc0 = 4 + 16 * (p % 3);
        v16h afrag;
        unsigned int* ap = (unsigned int*)&afrag;
        const unsigned int* rowp = (const unsigned int*)&outv[lr0 - 4 + n16][0];
        #pragma unroll
        for (int v = 0; v < 8; ++v) {
          int k = 2 * v + 16 * half;  // even
          ap[v] = rowp[(lc0 - 4 + k) >> 1];
        }
        v8f c = {};
        c = __builtin_amdgcn_wmma_f32_16x16x32_f16(false, afrag, false, bG,
                                                   (short)0, c, false, false);
        #pragma unroll
        for (int j = 0; j < 8; ++j) {
          int lr = lr0 + j + 8 * half;
          int lc = lc0 + n16;
          int gr = ty + lr - 6, gc = tx + lc - 6;
          // zero blur outside image: matches reference per-stage zero padding
          float val = (gr >= 0 && gr < IMG && gc >= 0 && gc < IMG) ? c[j] : 0.f;
          blur[lr - 4][lc - 4] = val;
        }
      }
      __syncthreads();

      // ---- stage 3: Sobel + magnitude/orientation accumulation (VALU) ----
      for (int i = tid; i < 34 * 34; i += 256) {
        int lr = 5 + i / 34;
        int lc = 5 + i % 34;
        int r = lr - 4, c = lc - 4;
        float a0 = blur[r - 1][c - 1], a1 = blur[r - 1][c], a2 = blur[r - 1][c + 1];
        float b0 = blur[r][c - 1],                          b2 = blur[r][c + 1];
        float d0 = blur[r + 1][c - 1], d1 = blur[r + 1][c], d2 = blur[r + 1][c + 1];
        float gx = (a0 - a2) + 2.f * (b0 - b2) + (d0 - d2);
        float gy = (a0 - d0) + 2.f * (a1 - d1) + (a2 - d2);
        mag[r][c] += sqrtf(gx * gx + gy * gy);
        gxs[r][c] += gx;
        gys[r][c] += gy;
      }
      __syncthreads();
    } // channels

    // ---- stage 4: orientation NMS + threshold + L1 ----
    for (int i = tid; i < 32 * 32; i += 256) {
      int r = i >> 5, c = i & 31;
      float cm  = mag[r + 2][c + 2];
      float gxv = gxs[r + 2][c + 2];
      float gyv = gys[r + 2][c + 2];
      float ang = atan2f(gyv, gxv) * 57.29577951308232f + 180.f;
      int k  = ((int)rintf(ang * (1.f / 45.f))) & 7;   // rint = half-even = jnp.round
      int kn = (k + 4) & 7;
      int gr = ty + r, gc = tx + c;
      int n1r = gr + offy[k],  n1c = gc + offx[k];
      int n2r = gr + offy[kn], n2c = gc + offx[kn];
      float m1 = (n1r >= 0 && n1r < IMG && n1c >= 0 && n1c < IMG)
                     ? mag[r + 2 + offy[k]][c + 2 + offx[k]] : 0.f;
      float m2 = (n2r >= 0 && n2r < IMG && n2c >= 0 && n2c < IMG)
                     ? mag[r + 2 + offy[kn]][c + 2 + offx[kn]] : 0.f;
      float pos = cm - m1, neg = cm - m2;
      float thin = (fminf(pos, neg) > 0.f) ? cm : 0.f;
      float e = (thin < 2.f) ? 0.f : thin;
      if (imgSel == 0) e0[r][c] = e;
      else             acc += fabsf(e - e0[r][c]);
    }
    __syncthreads();
  } // imgSel

  // ---- deterministic block reduction (red aliases blur; blur is dead) ----
  red[tid] = acc;
  __syncthreads();
  for (int s = 128; s > 0; s >>= 1) {
    if (tid < s) red[tid] += red[tid + s];
    __syncthreads();
  }
  if (tid == 0) partial[bx] = red[0];
}

__global__ __launch_bounds__(256)
void reduce_partials(const float* __restrict__ partial, float* __restrict__ outp, int n)
{
  __shared__ float red[256];
  float a = 0.f;
  for (int i = threadIdx.x; i < n; i += 256) a += partial[i];
  red[threadIdx.x] = a;
  __syncthreads();
  for (int s = 128; s > 0; s >>= 1) {
    if (threadIdx.x < s) red[threadIdx.x] += red[threadIdx.x + s];
    __syncthreads();
  }
  if (threadIdx.x == 0) outp[0] = red[0] * (1.0f / 4194304.0f); // mean over 64*256*256
}

extern "C" void kernel_launch(void* const* d_in, const int* in_sizes, int n_in,
                              void* d_out, int out_size, void* d_ws, size_t ws_size,
                              hipStream_t stream) {
  const float* tgt = (const float*)d_in[0];   // data_input
  const float* mdl = (const float*)d_in[1];   // model_output
  float* partial = (float*)d_ws;              // 4096 floats
  canny_l1_tile_kernel<<<4096, 256, 0, stream>>>(tgt, mdl, partial);
  reduce_partials<<<1, 256, 0, stream>>>(partial, (float*)d_out, 4096);
}
